// BoundaryAwareViT_74500502716591
// MI455X (gfx1250) — compile-verified, matrix operand order checked by reading, not run on hardware
//
#include <hip/hip_runtime.h>
#include <cmath>

// ---------------------------------------------------------------------------
// BoundaryAwareViT forward for MI455X (gfx1250), wave32 + WMMA bf16.
// ---------------------------------------------------------------------------

typedef __bf16 bf16_t;
typedef __attribute__((ext_vector_type(16))) __bf16 v16bf;
typedef __attribute__((ext_vector_type(8)))  __bf16 v8bf;
typedef __attribute__((ext_vector_type(8)))  float  v8f;

#define Bb    32
#define Gg    32
#define NTOK  1024
#define BN    32768       // Bb * NTOK
#define Dd    256
#define DQq   32
#define DFf   1024
#define NDEPTH 8
#define SCALE_QK 0.17677669529663687f   // 1/sqrt(32)

// ---------------------------------------------------------------------------
// WMMA fragment loaders (CDNA5 ISA 7.12.2 layouts, wave32)
// A (16x32 MxK, 16-bit): lane m = lane%16, kb = (lane/16)*8,
//   elements [0..7]  = A[m, k0+kb+0..7]
//   elements [8..15] = A[m, k0+kb+16+0..7]
// B (32x16 KxN) with storage [n, k] row-major uses the identical pattern.
// ---------------------------------------------------------------------------
__device__ inline v16bf load_frag16(const bf16_t* __restrict__ base, int ld,
                                    int r0, int k0) {
  const int lane = threadIdx.x & 31;
  const int r = lane & 15;
  const int kb = (lane >> 4) << 3;
  const bf16_t* p = base + (size_t)(r0 + r) * (size_t)ld + (size_t)(k0 + kb);
  v8bf lo = *(const v8bf*)p;           // global_load_b128
  v8bf hi = *(const v8bf*)(p + 16);    // global_load_b128
  v16bf out;
#pragma unroll
  for (int i = 0; i < 8; ++i) { out[i] = lo[i]; out[i + 8] = hi[i]; }
  return out;
}

// B operand stored naturally as [k, n] (row stride ld): gather per-lane column.
__device__ inline v16bf load_frag16_col(const bf16_t* __restrict__ base, int ld,
                                        int n0, int k0) {
  const int lane = threadIdx.x & 31;
  const int n = n0 + (lane & 15);
  const int kb = k0 + ((lane >> 4) << 3);
  v16bf out;
#pragma unroll
  for (int i = 0; i < 8; ++i) {
    out[i]     = base[(size_t)(kb + i) * (size_t)ld + n];
    out[i + 8] = base[(size_t)(kb + 16 + i) * (size_t)ld + n];
  }
  return out;
}

// ---------------------------------------------------------------------------
// Register-blocked GEMM: C[M,N] = epilogue(A[M,K] @ Wt[N,K]^T + bias)
// block = 128 threads = 4 waves arranged 2(M) x 2(N).
// Each wave owns a WM x WN grid of 16x16 accumulators (8 WMMAs per K-step at
// WM=2,WN=4, fed by 6 fragment loads -> ~2.7x better WMMA/byte than 1x1).
// ---------------------------------------------------------------------------
#define EPI_PATCH    0   // f32 out = acc + bias + pos
#define EPI_BIAS_BF  1   // bf16 out = acc + bias
#define EPI_TANH_ADD 2   // f32 out += tanh(acc + bias)
#define EPI_GELU_BF  3   // bf16 out = gelu_exact(acc + bias)
#define EPI_BIAS_ADD 4   // f32 out += acc + bias

template <int EPI, int WM, int WN>
__global__ __launch_bounds__(128) void gemm_bf16_kernel(
    const bf16_t* __restrict__ A, const bf16_t* __restrict__ Wt,
    const float* __restrict__ bias, int K, int N, int nTilesN,
    float* __restrict__ outf, bf16_t* __restrict__ outb,
    const float* __restrict__ pos) {
  const int wave = threadIdx.x >> 5;
  const int lane = threadIdx.x & 31;
  const int mt = blockIdx.x / nTilesN;
  const int nt = blockIdx.x % nTilesN;
  const int m0 = mt * (32 * WM) + (wave >> 1) * (16 * WM);
  const int n0 = nt * (32 * WN) + (wave & 1) * (16 * WN);

  v8f acc[WM][WN];
#pragma unroll
  for (int i = 0; i < WM; ++i)
#pragma unroll
    for (int j = 0; j < WN; ++j) acc[i][j] = (v8f){};

  for (int k0 = 0; k0 < K; k0 += 32) {
    v16bf af[WM];
    v16bf bf[WN];
#pragma unroll
    for (int i = 0; i < WM; ++i) af[i] = load_frag16(A, K, m0 + i * 16, k0);
#pragma unroll
    for (int j = 0; j < WN; ++j) bf[j] = load_frag16(Wt, K, n0 + j * 16, k0);
    if (k0 + 32 < K) {  // engage global_prefetch for the next K chunk
      __builtin_prefetch(A + (size_t)m0 * K + k0 + 32, 0, 1);
      __builtin_prefetch(Wt + (size_t)n0 * K + k0 + 32, 0, 1);
    }
#pragma unroll
    for (int i = 0; i < WM; ++i)
#pragma unroll
      for (int j = 0; j < WN; ++j)
        acc[i][j] = __builtin_amdgcn_wmma_f32_16x16x32_bf16(
            false, af[i], false, bf[j], (short)0, acc[i][j], false, false);
  }

  const int nl = lane & 15;
  const int mh = (lane >> 4) << 3;
#pragma unroll
  for (int i = 0; i < WM; ++i) {
#pragma unroll
    for (int j = 0; j < WN; ++j) {
      const int n = n0 + j * 16 + nl;
      const int mb = m0 + i * 16 + mh;
      const float bn = bias[n];
#pragma unroll
      for (int r = 0; r < 8; ++r) {
        const int m = mb + r;
        const size_t oidx = (size_t)m * (size_t)N + n;
        float v = acc[i][j][r] + bn;
        if (EPI == EPI_PATCH) {
          outf[oidx] = v + pos[(size_t)(m & (NTOK - 1)) * Dd + n];
        } else if (EPI == EPI_BIAS_BF) {
          outb[oidx] = (bf16_t)v;
        } else if (EPI == EPI_TANH_ADD) {
          outf[oidx] += tanhf(v);
        } else if (EPI == EPI_GELU_BF) {
          outb[oidx] =
              (bf16_t)(0.5f * v * (1.0f + erff(v * 0.7071067811865476f)));
        } else {
          outf[oidx] += v;
        }
      }
    }
  }
}

// ---------------------------------------------------------------------------
// im2col for patch embed: A[token, py*16+px] = x[b, gy*16+py, gx*16+px]
// ---------------------------------------------------------------------------
__global__ __launch_bounds__(256) void im2col_kernel(
    const float* __restrict__ x, bf16_t* __restrict__ A) {
  const size_t idx = (size_t)blockIdx.x * 256 + threadIdx.x;  // BN*256 total
  const int k = (int)(idx & 255);
  const size_t tok = idx >> 8;
  const int b = (int)(tok >> 10);
  const int n = (int)(tok & 1023);
  const int gy = n >> 5, gx = n & 31;
  const int py = k >> 4, px = k & 15;
  const size_t xi = ((size_t)b * 512 + (size_t)(gy * 16 + py)) * 512 +
                    (size_t)(gx * 16 + px);
  A[idx] = (bf16_t)x[xi];
}

// ---------------------------------------------------------------------------
// 5-point Laplacian over token grid (SAME, zero pad) -> bf16
// ---------------------------------------------------------------------------
__global__ __launch_bounds__(256) void lap_kernel(
    const float* __restrict__ h, bf16_t* __restrict__ outb) {
  const size_t idx = (size_t)blockIdx.x * 256 + threadIdx.x;  // BN*Dd total
  const size_t tok = idx >> 8;
  const int n = (int)(tok & 1023);
  const int gy = n >> 5, gx = n & 31;
  const float c = h[idx];
  const float up = (gy > 0)      ? h[idx - (size_t)Gg * Dd] : 0.0f;
  const float dn = (gy < Gg - 1) ? h[idx + (size_t)Gg * Dd] : 0.0f;
  const float lf = (gx > 0)      ? h[idx - Dd] : 0.0f;
  const float rt = (gx < Gg - 1) ? h[idx + Dd] : 0.0f;
  outb[idx] = (bf16_t)(4.0f * c - up - dn - lf - rt);
}

// ---------------------------------------------------------------------------
// LayerNorm over D=256: one wave per token, 8 elems/lane, shfl reduction
// ---------------------------------------------------------------------------
__global__ __launch_bounds__(256) void ln_kernel(
    const float* __restrict__ h, const float* __restrict__ g,
    const float* __restrict__ bta, bf16_t* __restrict__ xn) {
  const int wave = threadIdx.x >> 5;
  const int lane = threadIdx.x & 31;
  const size_t tok = (size_t)blockIdx.x * 8 + wave;
  const float* p = h + tok * Dd + lane * 8;
  float v[8];
  float s = 0.0f;
#pragma unroll
  for (int i = 0; i < 8; ++i) { v[i] = p[i]; s += v[i]; }
#pragma unroll
  for (int off = 16; off > 0; off >>= 1) s += __shfl_xor(s, off, 32);
  const float mean = s * (1.0f / Dd);
  float q = 0.0f;
#pragma unroll
  for (int i = 0; i < 8; ++i) { const float d = v[i] - mean; q += d * d; }
#pragma unroll
  for (int off = 16; off > 0; off >>= 1) q += __shfl_xor(q, off, 32);
  const float rstd = rsqrtf(q * (1.0f / Dd) + 1e-5f);
#pragma unroll
  for (int i = 0; i < 8; ++i) {
    const int c = lane * 8 + i;
    xn[tok * Dd + c] = (bf16_t)((v[i] - mean) * rstd * g[c] + bta[c]);
  }
}

// ---------------------------------------------------------------------------
// Criss-cross attention scores. Row: per (b,h) block; Col: per (b,w) block.
// 32x32 output per block, K=DQ=32 -> single WMMA per wave (4 waves).
// ---------------------------------------------------------------------------
__global__ __launch_bounds__(128) void row_scores_kernel(
    const bf16_t* __restrict__ q, const bf16_t* __restrict__ k,
    float* __restrict__ scores) {
  const int b = blockIdx.x >> 5, h = blockIdx.x & 31;
  const int wave = threadIdx.x >> 5, lane = threadIdx.x & 31;
  const size_t tokbase = ((size_t)b * Gg + h) * Gg;
  const bf16_t* Q = q + tokbase * DQq;   // rows over w, lda = DQ
  const bf16_t* Kp = k + tokbase * DQq;
  const int m0 = (wave >> 1) * 16, n0 = (wave & 1) * 16;
  v16bf af = load_frag16(Q, DQq, m0, 0);
  v16bf bf = load_frag16(Kp, DQq, n0, 0);
  v8f acc = {};
  acc = __builtin_amdgcn_wmma_f32_16x16x32_bf16(false, af, false, bf,
                                                (short)0, acc, false, false);
  const int n = n0 + (lane & 15);
  const int mb = m0 + ((lane >> 4) << 3);
#pragma unroll
  for (int r = 0; r < 8; ++r)
    scores[(tokbase + mb + r) * 64 + n] = acc[r] * SCALE_QK;
}

__global__ __launch_bounds__(128) void col_scores_kernel(
    const bf16_t* __restrict__ q, const bf16_t* __restrict__ k,
    float* __restrict__ scores) {
  const int b = blockIdx.x >> 5, w = blockIdx.x & 31;
  const int wave = threadIdx.x >> 5, lane = threadIdx.x & 31;
  const bf16_t* Q = q + ((size_t)b * Gg * Gg + w) * DQq;   // rows over h
  const bf16_t* Kp = k + ((size_t)b * Gg * Gg + w) * DQq;  // rows over u
  const int m0 = (wave >> 1) * 16, n0 = (wave & 1) * 16;
  v16bf af = load_frag16(Q, Gg * DQq, m0, 0);
  v16bf bf = load_frag16(Kp, Gg * DQq, n0, 0);
  v8f acc = {};
  acc = __builtin_amdgcn_wmma_f32_16x16x32_bf16(false, af, false, bf,
                                                (short)0, acc, false, false);
  const int n = n0 + (lane & 15);  // u
  const int mb = m0 + ((lane >> 4) << 3);
#pragma unroll
  for (int r = 0; r < 8; ++r) {
    const int hh = mb + r;
    float v = acc[r] * SCALE_QK;
    if (n == hh) v -= 1e9f;  // CCNet duplicated-self mask
    const size_t tok = ((size_t)b * Gg + hh) * Gg + w;
    scores[tok * 64 + 32 + n] = v;
  }
}

// softmax over 64 scores per token -> bf16 probs; one wave per token
__global__ __launch_bounds__(256) void softmax_kernel(
    const float* __restrict__ scores, bf16_t* __restrict__ attn) {
  const int wave = threadIdx.x >> 5, lane = threadIdx.x & 31;
  const size_t tok = (size_t)blockIdx.x * 8 + wave;
  const float s0 = scores[tok * 64 + lane];
  const float s1 = scores[tok * 64 + 32 + lane];
  float mx = fmaxf(s0, s1);
#pragma unroll
  for (int off = 16; off > 0; off >>= 1) mx = fmaxf(mx, __shfl_xor(mx, off, 32));
  const float e0 = expf(s0 - mx);
  const float e1 = expf(s1 - mx);
  float sum = e0 + e1;
#pragma unroll
  for (int off = 16; off > 0; off >>= 1) sum += __shfl_xor(sum, off, 32);
  const float inv = 1.0f / sum;
  attn[tok * 64 + lane] = (bf16_t)(e0 * inv);
  attn[tok * 64 + 32 + lane] = (bf16_t)(e1 * inv);
}

// Row apply: out_row[w, c] = sum_v a_row[w,v] * V[b,h,v,c]  (per (b,h))
__global__ __launch_bounds__(128) void attn_apply_row_kernel(
    const bf16_t* __restrict__ attn, const bf16_t* __restrict__ v,
    float* __restrict__ attn_out) {
  const int b = blockIdx.x >> 5, h = blockIdx.x & 31;
  const int wave = threadIdx.x >> 5, lane = threadIdx.x & 31;
  const size_t tokbase = ((size_t)b * Gg + h) * Gg;
  const bf16_t* Abase = attn + tokbase * 64;  // rows over w, cols 0..31
  const bf16_t* Vbase = v + tokbase * Dd;     // [vtok, c], ld = D
  const int m0 = (wave >> 1) * 16;
  v16bf af = load_frag16(Abase, 64, m0, 0);
  const int nstart = (wave & 1) * 128;
  for (int ntile = 0; ntile < 8; ++ntile) {
    const int n0 = nstart + ntile * 16;
    v16bf bf = load_frag16_col(Vbase, Dd, n0, 0);
    v8f acc = {};
    acc = __builtin_amdgcn_wmma_f32_16x16x32_bf16(false, af, false, bf,
                                                  (short)0, acc, false, false);
    const int n = n0 + (lane & 15);
    const int mb = m0 + ((lane >> 4) << 3);
#pragma unroll
    for (int r = 0; r < 8; ++r)
      attn_out[(tokbase + mb + r) * Dd + n] = acc[r];
  }
}

// Col apply + residual fuse: h += gamma*(row_part + col_part) + xn  (per (b,w))
__global__ __launch_bounds__(128) void attn_apply_col_kernel(
    const bf16_t* __restrict__ attn, const bf16_t* __restrict__ v,
    const float* __restrict__ attn_out, const bf16_t* __restrict__ xn,
    const float* __restrict__ gamma, int depth, float* __restrict__ h) {
  const int b = blockIdx.x >> 5, w = blockIdx.x & 31;
  const int wave = threadIdx.x >> 5, lane = threadIdx.x & 31;
  const float gm = gamma[depth];
  const bf16_t* Abase = attn + ((size_t)b * Gg * Gg + w) * 64 + 32; // rows over h
  const bf16_t* Vbase = v + ((size_t)b * Gg * Gg + w) * Dd;         // [u, c]
  const int m0 = (wave >> 1) * 16;
  v16bf af = load_frag16(Abase, Gg * 64, m0, 0);
  const int nstart = (wave & 1) * 128;
  for (int ntile = 0; ntile < 8; ++ntile) {
    const int n0 = nstart + ntile * 16;
    v16bf bf = load_frag16_col(Vbase, Gg * Dd, n0, 0);
    v8f acc = {};
    acc = __builtin_amdgcn_wmma_f32_16x16x32_bf16(false, af, false, bf,
                                                  (short)0, acc, false, false);
    const int n = n0 + (lane & 15);
    const int mb = m0 + ((lane >> 4) << 3);
#pragma unroll
    for (int r = 0; r < 8; ++r) {
      const int hh = mb + r;
      const size_t tok = ((size_t)b * Gg + hh) * Gg + w;
      const size_t oidx = tok * Dd + n;
      h[oidx] += gm * (attn_out[oidx] + acc[r]) + (float)xn[oidx];
    }
  }
}

// Head: out[b,1,gy,gx] = h[token,:] . w_head + b_head   (token order == flat out)
__global__ __launch_bounds__(256) void head_kernel(
    const float* __restrict__ h, const float* __restrict__ w,
    const float* __restrict__ bh, float* __restrict__ out) {
  const int wave = threadIdx.x >> 5, lane = threadIdx.x & 31;
  const size_t tok = (size_t)blockIdx.x * 8 + wave;
  float s = 0.0f;
#pragma unroll
  for (int i = 0; i < 8; ++i)
    s += h[tok * Dd + lane * 8 + i] * w[lane * 8 + i];
#pragma unroll
  for (int off = 16; off > 0; off >>= 1) s += __shfl_xor(s, off, 32);
  if (lane == 0) out[tok] = s + bh[0];
}

// Weight prep: convert, and transpose-convert [K,N] f32 -> [N,K] bf16
__global__ __launch_bounds__(256) void cvt_kernel(
    const float* __restrict__ src, bf16_t* __restrict__ dst, int count) {
  const int i = blockIdx.x * 256 + threadIdx.x;
  if (i < count) dst[i] = (bf16_t)src[i];
}
__global__ __launch_bounds__(256) void tconv_kernel(
    const float* __restrict__ src, bf16_t* __restrict__ dst, int K, int N) {
  const int i = blockIdx.x * 256 + threadIdx.x;
  if (i < K * N) {
    const int k = i / N, n = i % N;
    dst[(size_t)n * K + k] = src[i];
  }
}

// ---------------------------------------------------------------------------
extern "C" void kernel_launch(void* const* d_in, const int* in_sizes, int n_in,
                              void* d_out, int out_size, void* d_ws,
                              size_t ws_size, hipStream_t stream) {
  (void)in_sizes; (void)n_in; (void)out_size; (void)ws_size;
  const float* x       = (const float*)d_in[0];
  const float* w_patch = (const float*)d_in[1];
  const float* b_patch = (const float*)d_in[2];
  const float* pos     = (const float*)d_in[3];
  const float* w_edge  = (const float*)d_in[4];
  const float* b_edge  = (const float*)d_in[5];
  const float* ln_g    = (const float*)d_in[6];
  const float* ln_b    = (const float*)d_in[7];
  const float* wq      = (const float*)d_in[8];
  const float* bq      = (const float*)d_in[9];
  const float* wk      = (const float*)d_in[10];
  const float* bk      = (const float*)d_in[11];
  const float* wv      = (const float*)d_in[12];
  const float* bv      = (const float*)d_in[13];
  const float* gamma   = (const float*)d_in[14];
  const float* w1      = (const float*)d_in[15];
  const float* b1      = (const float*)d_in[16];
  const float* w2      = (const float*)d_in[17];
  const float* b2      = (const float*)d_in[18];
  const float* w_head  = (const float*)d_in[19];
  const float* b_head  = (const float*)d_in[20];

  // ---- workspace layout -------------------------------------------------
  char* base = (char*)d_ws;
  size_t off = 0;
  auto alloc = [&](size_t bytes) -> void* {
    void* p = base + off;
    off += (bytes + 255) & ~(size_t)255;
    return p;
  };
  float*  hbuf = (float*) alloc((size_t)BN * Dd * 4);   // residual stream
  bf16_t* xnb  = (bf16_t*)alloc((size_t)BN * Dd * 2);   // LN output (bf16)
  bf16_t* abuf = (bf16_t*)alloc((size_t)BN * 256 * 2);  // im2col / laplacian
  // union region (64 MB): attention buffers alias the MLP mid buffer
  char* ub = (char*)alloc((size_t)BN * DFf * 2);
  bf16_t* qb = (bf16_t*)(ub);
  bf16_t* kb = (bf16_t*)(ub + (size_t)BN * DQq * 2);
  bf16_t* vb = (bf16_t*)(ub + (size_t)BN * DQq * 4);
  float*  sc = (float*) (ub + (size_t)BN * DQq * 4 + (size_t)BN * Dd * 2);
  bf16_t* at = (bf16_t*)(ub + (size_t)BN * DQq * 4 + (size_t)BN * Dd * 2 +
                         (size_t)BN * 64 * 4);
  float*  ao = (float*) (ub + (size_t)BN * DQq * 4 + (size_t)BN * Dd * 2 +
                         (size_t)BN * 64 * 4 + (size_t)BN * 64 * 2);
  bf16_t* mid = (bf16_t*)ub;                            // [BN, DF] bf16
  // bf16 weights ([out, in] layout for the B fragment)
  bf16_t* wpT = (bf16_t*)alloc((size_t)Dd * 256 * 2);
  bf16_t* weT = (bf16_t*)alloc((size_t)Dd * Dd * 2);
  bf16_t* wqT = (bf16_t*)alloc((size_t)NDEPTH * DQq * Dd * 2);
  bf16_t* wkT = (bf16_t*)alloc((size_t)NDEPTH * DQq * Dd * 2);
  bf16_t* wvT = (bf16_t*)alloc((size_t)NDEPTH * Dd * Dd * 2);
  bf16_t* w1T = (bf16_t*)alloc((size_t)NDEPTH * DFf * Dd * 2);
  bf16_t* w2T = (bf16_t*)alloc((size_t)NDEPTH * Dd * DFf * 2);

  // ---- weight prep ------------------------------------------------------
  cvt_kernel<<<(Dd * 256) / 256, 256, 0, stream>>>(w_patch, wpT, Dd * 256);
  tconv_kernel<<<(Dd * Dd) / 256, 256, 0, stream>>>(w_edge, weT, Dd, Dd);
  for (int d = 0; d < NDEPTH; ++d) {
    tconv_kernel<<<(Dd * DQq) / 256, 256, 0, stream>>>(
        wq + (size_t)d * Dd * DQq, wqT + (size_t)d * DQq * Dd, Dd, DQq);
    tconv_kernel<<<(Dd * DQq) / 256, 256, 0, stream>>>(
        wk + (size_t)d * Dd * DQq, wkT + (size_t)d * DQq * Dd, Dd, DQq);
    tconv_kernel<<<(Dd * Dd) / 256, 256, 0, stream>>>(
        wv + (size_t)d * Dd * Dd, wvT + (size_t)d * Dd * Dd, Dd, Dd);
    tconv_kernel<<<(Dd * DFf) / 256, 256, 0, stream>>>(
        w1 + (size_t)d * Dd * DFf, w1T + (size_t)d * DFf * Dd, Dd, DFf);
    tconv_kernel<<<(DFf * Dd) / 256, 256, 0, stream>>>(
        w2 + (size_t)d * DFf * Dd, w2T + (size_t)d * Dd * DFf, DFf, Dd);
  }

  // ---- patch embed + pos (M=BN, N=256, block tile 64x128) ---------------
  im2col_kernel<<<(BN * 256) / 256, 256, 0, stream>>>(x, abuf);
  gemm_bf16_kernel<EPI_PATCH, 2, 4><<<(BN / 64) * (Dd / 128), 128, 0, stream>>>(
      abuf, wpT, b_patch, 256, Dd, Dd / 128, hbuf, nullptr, pos);

  // ---- edge tokens ------------------------------------------------------
  lap_kernel<<<(BN * Dd) / 256, 256, 0, stream>>>(hbuf, abuf);
  gemm_bf16_kernel<EPI_TANH_ADD, 2, 4>
      <<<(BN / 64) * (Dd / 128), 128, 0, stream>>>(
          abuf, weT, b_edge, Dd, Dd, Dd / 128, hbuf, nullptr, nullptr);

  // ---- transformer blocks ----------------------------------------------
  for (int d = 0; d < NDEPTH; ++d) {
    const float* lg = ln_g + (size_t)d * Dd;
    const float* lb = ln_b + (size_t)d * Dd;

    // attention sublayer
    ln_kernel<<<BN / 8, 256, 0, stream>>>(hbuf, lg, lb, xnb);
    gemm_bf16_kernel<EPI_BIAS_BF, 2, 1>
        <<<(BN / 64) * (DQq / 32), 128, 0, stream>>>(
            xnb, wqT + (size_t)d * DQq * Dd, bq + (size_t)d * DQq, Dd, DQq,
            DQq / 32, nullptr, qb, nullptr);
    gemm_bf16_kernel<EPI_BIAS_BF, 2, 1>
        <<<(BN / 64) * (DQq / 32), 128, 0, stream>>>(
            xnb, wkT + (size_t)d * DQq * Dd, bk + (size_t)d * DQq, Dd, DQq,
            DQq / 32, nullptr, kb, nullptr);
    gemm_bf16_kernel<EPI_BIAS_BF, 2, 4>
        <<<(BN / 64) * (Dd / 128), 128, 0, stream>>>(
            xnb, wvT + (size_t)d * Dd * Dd, bv + (size_t)d * Dd, Dd, Dd,
            Dd / 128, nullptr, vb, nullptr);
    row_scores_kernel<<<Bb * Gg, 128, 0, stream>>>(qb, kb, sc);
    col_scores_kernel<<<Bb * Gg, 128, 0, stream>>>(qb, kb, sc);
    softmax_kernel<<<BN / 8, 256, 0, stream>>>(sc, at);
    attn_apply_row_kernel<<<Bb * Gg, 128, 0, stream>>>(at, vb, ao);
    attn_apply_col_kernel<<<Bb * Gg, 128, 0, stream>>>(at, vb, ao, xnb, gamma,
                                                       d, hbuf);

    // MLP sublayer (same LN params, per reference)
    ln_kernel<<<BN / 8, 256, 0, stream>>>(hbuf, lg, lb, xnb);
    gemm_bf16_kernel<EPI_GELU_BF, 2, 4>
        <<<(BN / 64) * (DFf / 128), 128, 0, stream>>>(
            xnb, w1T + (size_t)d * DFf * Dd, b1 + (size_t)d * DFf, Dd, DFf,
            DFf / 128, nullptr, mid, nullptr);
    gemm_bf16_kernel<EPI_BIAS_ADD, 2, 4>
        <<<(BN / 64) * (Dd / 128), 128, 0, stream>>>(
            mid, w2T + (size_t)d * Dd * DFf, b2 + (size_t)d * Dd, DFf, Dd,
            Dd / 128, hbuf, nullptr, nullptr);
  }

  // ---- head -------------------------------------------------------------
  head_kernel<<<BN / 8, 256, 0, stream>>>(hbuf, w_head, b_head, (float*)d_out);
}
